// GenerativeMPS_58574763983328
// MI455X (gfx1250) — matrix operand order repaired and checked by reading, not compiled, not run on hardware
//
#include <hip/hip_runtime.h>

// MPS generative-model evaluation for MI455X (gfx1250, wave32, WMMA).
//
// Roofline: mps = 134 MB (fits in 192 MB L2), ~51 GFLOP of matmul ->
// compute/latency bound, not HBM bound. Strategy: bf16 hi/lo split WMMA
// (3x v_wmma_f32_16x16x32_bf16 per 16x16x32 step ~ fp32 accuracy at
// ~2.7x fewer matrix ops than V_WMMA_F32_16X16X4_F32), per-site tiles
// staged transposed into LDS once and reused by all 32 waves.
// hi = bit-truncated bf16 (residual exact), lo = RNE bf16 (unbiased).

#define PI_2 1.57079632679489662f

constexpr int Nn = 1024;   // sites
constexpr int Dd = 128;    // bond dimension
constexpr int Bb = 512;    // batch
constexpr int MAIN_ROWS   = 64;
constexpr int MAIN_BLOCKS = Bb / MAIN_ROWS;  // 8
constexpr int SITE_FLOATS = Dd * Dd * 2;     // 32768 floats / site

typedef __attribute__((ext_vector_type(16))) __bf16 v16bf;
typedef __attribute__((ext_vector_type(8)))  float  v8f;

union Frag32 { uint4 u[2]; v16bf v; };

// RNE f32 -> bf16 (used for the lo plane only; keeps the split unbiased).
__device__ __forceinline__ __bf16 f2bf_rne(float f) {
  unsigned u = __builtin_bit_cast(unsigned, f);
  unsigned r = u + 0x7FFFu + ((u >> 16) & 1u);
  unsigned short h = (unsigned short)(r >> 16);
  return __builtin_bit_cast(__bf16, h);
}
// hi part of f as float (low 16 mantissa bits cleared) — residual is exact.
__device__ __forceinline__ float bfhi_f(float f) {
  return __builtin_bit_cast(float, __builtin_bit_cast(unsigned, f) & 0xFFFF0000u);
}
__device__ __forceinline__ __bf16 bf_hi(float f) {  // truncated hi bf16
  return __builtin_bit_cast(__bf16, (unsigned short)(__builtin_bit_cast(unsigned, f) >> 16));
}
__device__ __forceinline__ __bf16 bf_lo(float f) {  // RNE residual bf16
  return f2bf_rne(f - bfhi_f(f));
}
// pack top-16 of two floats into one dword (v_perm_b32 candidate)
__device__ __forceinline__ unsigned pack_hi2(float a, float b) {
  unsigned ua = __builtin_bit_cast(unsigned, a);
  unsigned ub = __builtin_bit_cast(unsigned, b);
  return (ua >> 16) | (ub & 0xFFFF0000u);
}
__device__ __forceinline__ unsigned pack_lo2(float a, float b) {
  return (unsigned)__builtin_bit_cast(unsigned short, bf_lo(a)) |
         ((unsigned)__builtin_bit_cast(unsigned short, bf_lo(b)) << 16);
}

__device__ __forceinline__ v8f vzero() {
  v8f z;
#pragma unroll
  for (int i = 0; i < 8; ++i) z[i] = 0.f;
  return z;
}

// A-fragment (M=16 x K=32 bf16): per lane 16 contiguous K (32 contiguous bytes).
__device__ __forceinline__ v16bf fragA(const __bf16* p) {
  Frag32 f;
  f.u[0] = *(const uint4*)p;
  f.u[1] = *(const uint4*)(p + 8);
  return f.v;
}
// B-fragment (K=32 x N=16 bf16) from transposed LDS store: 32 contiguous bytes.
__device__ __forceinline__ v16bf fragB(const __bf16* p) {
  Frag32 f;
  f.u[0] = *(const uint4*)p;
  f.u[1] = *(const uint4*)(p + 8);
  return f.v;
}

// hi/lo 3-product accumulate: (ah+al)(bh+bl) ~= ah*bh + ah*bl + al*bh
__device__ __forceinline__ v8f wmma3(v16bf ah, v16bf al, v16bf bh, v16bf bl, v8f acc) {
  acc = __builtin_amdgcn_wmma_f32_16x16x32_bf16(false, ah, false, bh, (short)0, acc, false, false);
  acc = __builtin_amdgcn_wmma_f32_16x16x32_bf16(false, ah, false, bl, (short)0, acc, false, false);
  acc = __builtin_amdgcn_wmma_f32_16x16x32_bf16(false, al, false, bh, (short)0, acc, false, false);
  return acc;
}

// Store 4 consecutive-l hi/lo values (one b64 each) into transposed planes.
__device__ __forceinline__ void store_quad_hl(__bf16* Thi, __bf16* Tlo, int off,
                                              float f0, float f1, float f2, float f3) {
  uint2 hw, lw;
  hw.x = pack_hi2(f0, f1);
  hw.y = pack_hi2(f2, f3);
  lw.x = pack_lo2(f0, f1);
  lw.y = pack_lo2(f2, f3);
  *(uint2*)(Thi + off) = hw;
  *(uint2*)(Tlo + off) = lw;
}

// Stage one site's A (layout [l][r][c], c fastest, fp32) into LDS as
// transposed bf16 hi/lo planes T[c][r][l]. 1024 threads, 8 float4 loads each.
__device__ __forceinline__ void stage_site(const float* __restrict__ site,
                                           __bf16* Thi, __bf16* Tlo, int tid) {
  const float4* s4 = (const float4*)site;
#pragma unroll
  for (int i = 0; i < 2; ++i) {
    int p  = tid + i * 1024;                 // quad index in [0, 2048)
    int v  = ((p >> 6) << 8) | (p & 63);     // float4 index
    int l0 = (p >> 6) << 2;                  // 4 consecutive l values
    int r0 = (p & 63) << 1;                  // r, r+1
    float4 q0 = s4[v], q1 = s4[v + 64], q2 = s4[v + 128], q3 = s4[v + 192];
    int b00 = 0 * Dd * Dd + r0 * Dd + l0;
    int b10 = 1 * Dd * Dd + r0 * Dd + l0;
    store_quad_hl(Thi, Tlo, b00,      q0.x, q1.x, q2.x, q3.x);  // A0[l][r0]
    store_quad_hl(Thi, Tlo, b10,      q0.y, q1.y, q2.y, q3.y);  // A1[l][r0]
    store_quad_hl(Thi, Tlo, b00 + Dd, q0.z, q1.z, q2.z, q3.z);  // A0[l][r0+1]
    store_quad_hl(Thi, Tlo, b10 + Dd, q0.w, q1.w, q2.w, q3.w);  // A1[l][r0+1]
  }
}

__global__ __launch_bounds__(1024) void mps_kernel(const float* __restrict__ x,
                                                   const float* __restrict__ mps,
                                                   float* __restrict__ ws) {
  extern __shared__ __align__(16) char smem[];
  const int tid  = threadIdx.x;
  const int lane = tid & 31;
  const int wave = tid >> 5;
  const int l15  = lane & 15;
  const int hlf  = lane >> 4;

  __bf16* Thi = (__bf16*)smem;                 // [2][128][128] bf16 hi (64 KB)
  __bf16* Tlo = (__bf16*)(smem + 65536);       // lo plane (64 KB)

  if (blockIdx.x == 0) {
    // ======================== norm chain (single block) =====================
    __bf16* UThi = (__bf16*)(smem + 131072);   // U transposed hi (32 KB)
    __bf16* UTlo = (__bf16*)(smem + 163840);   // lo (32 KB)
    __bf16* ALhi = (__bf16*)(smem + 196608);   // Al hi (32 KB)
    __bf16* ALlo = (__bf16*)(smem + 229376);   // Al lo (32 KB)
    float*  Alf  = (float*)(smem + 131072);    // final f32 Al (reuses U region)

    // init Al = A0 @ A0^T, A0 = mps[0,0] (D,2) at flat r*2+c
    {
      const float* m00 = mps;
      int r = tid >> 3;
      int sb = (tid & 7) << 4;
      float a0 = m00[2 * r], a1 = m00[2 * r + 1];
#pragma unroll
      for (int j = 0; j < 16; ++j) {
        int s = sb + j;
        float val = a0 * m00[2 * s] + a1 * m00[2 * s + 1];
        ALhi[r * Dd + s] = bf_hi(val);
        ALlo[r * Dd + s] = bf_lo(val);
      }
    }
    stage_site(mps + (size_t)1 * SITE_FLOATS, Thi, Tlo, tid);
    __syncthreads();

    v8f acc2[2];
    for (int n = 1; n <= Nn - 2; ++n) {
      acc2[0] = vzero();
      acc2[1] = vzero();
#pragma unroll 1
      for (int c = 0; c < 2; ++c) {
        const __bf16* Tch = Thi + c * Dd * Dd;
        const __bf16* Tcl = Tlo + c * Dd * Dd;
        // stage 1: U_c = Al @ A_c (2 tiles/wave), write transposed hi/lo
#pragma unroll
        for (int j = 0; j < 2; ++j) {
          int t = wave * 2 + j, mt = t >> 3, nt = t & 7;
          v8f acc = vzero();
#pragma unroll
          for (int ks = 0; ks < 4; ++ks) {
            const __bf16* pa  = ALhi + (mt * 16 + l15) * Dd + ks * 32 + hlf * 16;
            const __bf16* pal = ALlo + (mt * 16 + l15) * Dd + ks * 32 + hlf * 16;
            const __bf16* pb  = Tch  + (nt * 16 + l15) * Dd + ks * 32 + hlf * 16;
            const __bf16* pbl = Tcl  + (nt * 16 + l15) * Dd + ks * 32 + hlf * 16;
            acc = wmma3(fragA(pa), fragA(pal), fragB(pb), fragB(pbl), acc);
          }
          // UT[r][a]: lane column r = nt*16+l15, 8 consecutive a -> one b128/plane
          int off = (nt * 16 + l15) * Dd + mt * 16 + hlf * 8;
          uint4 hw, lw;
          hw.x = pack_hi2(acc[0], acc[1]); hw.y = pack_hi2(acc[2], acc[3]);
          hw.z = pack_hi2(acc[4], acc[5]); hw.w = pack_hi2(acc[6], acc[7]);
          lw.x = pack_lo2(acc[0], acc[1]); lw.y = pack_lo2(acc[2], acc[3]);
          lw.z = pack_lo2(acc[4], acc[5]); lw.w = pack_lo2(acc[6], acc[7]);
          *(uint4*)(UThi + off) = hw;
          *(uint4*)(UTlo + off) = lw;
        }
        __syncthreads();
        // stage 2: Al' += A_c^T @ U_c  (A_c^T[m][k] = T[c][m][k], contiguous)
#pragma unroll
        for (int j = 0; j < 2; ++j) {
          int t = wave * 2 + j, mt = t >> 3, nt = t & 7;
#pragma unroll
          for (int ks = 0; ks < 4; ++ks) {
            const __bf16* pa  = Tch  + (mt * 16 + l15) * Dd + ks * 32 + hlf * 16;
            const __bf16* pal = Tcl  + (mt * 16 + l15) * Dd + ks * 32 + hlf * 16;
            const __bf16* pb  = UThi + (nt * 16 + l15) * Dd + ks * 32 + hlf * 16;
            const __bf16* pbl = UTlo + (nt * 16 + l15) * Dd + ks * 32 + hlf * 16;
            acc2[j] = wmma3(fragA(pa), fragA(pal), fragB(pb), fragB(pbl), acc2[j]);
          }
        }
        __syncthreads();
      }
      // fused phase: writeback new Al (or final f32) + stage next site's T
#pragma unroll
      for (int j = 0; j < 2; ++j) {
        int t = wave * 2 + j, mt = t >> 3, nt = t & 7;
        int scol = nt * 16 + l15;
        int rb = mt * 16 + hlf * 8;
        if (n < Nn - 2) {
#pragma unroll
          for (int v2 = 0; v2 < 8; ++v2) {
            float val = acc2[j][v2];
            ALhi[(rb + v2) * Dd + scol] = bf_hi(val);
            ALlo[(rb + v2) * Dd + scol] = bf_lo(val);
          }
        } else {
#pragma unroll
          for (int v2 = 0; v2 < 8; ++v2)
            Alf[(rb + v2) * Dd + scol] = acc2[j][v2];
        }
      }
      if (n + 1 <= Nn - 2) {
        stage_site(mps + (size_t)(n + 1) * SITE_FLOATS, Thi, Tlo, tid);
        if (n + 2 <= Nn - 2)
          __builtin_prefetch((const char*)(mps + (size_t)(n + 2) * SITE_FLOATS) + tid * 128, 0, 1);
      }
      __syncthreads();
    }

    // epilogue: norm^2 = sum_{r,s} Al[r,s] * (An[r] . An[s]), An = mps[-1,:,0,:]
    float* AnL = (float*)(smem + 196608);  // 256 floats, reuses Al-hi region
    if (tid < 256) {
      int l = tid >> 1, c = tid & 1;
      AnL[tid] = mps[((size_t)(Nn - 1) * Dd + l) * (size_t)(Dd * 2) + c];
    }
    __syncthreads();
    float part = 0.f;
    {
      int r = tid >> 3;
      int sb = (tid & 7) << 4;
      float a0 = AnL[2 * r], a1 = AnL[2 * r + 1];
#pragma unroll
      for (int j = 0; j < 16; ++j) {
        int s = sb + j;
        part += Alf[r * Dd + s] * (a0 * AnL[2 * s] + a1 * AnL[2 * s + 1]);
      }
    }
    float* red = (float*)smem;  // reuse T region
    __syncthreads();
    red[tid] = part;
    __syncthreads();
    for (int off = 512; off > 0; off >>= 1) {
      if (tid < off) red[tid] += red[tid + off];
      __syncthreads();
    }
    if (tid == 0) ws[Bb] = red[0];
  } else {
    // ================== batched amplitude chain (8 blocks) ==================
    const int rowbase = (blockIdx.x - 1) * MAIN_ROWS;
    __bf16* ALhi = (__bf16*)(smem + 131072);  // [64][128] hi (16 KB)
    __bf16* ALlo = (__bf16*)(smem + 147456);  // lo (16 KB)
    float*  eBuf = (float*)(smem + 163840);   // [2 parity][cos|sin][64]
    float*  Alf  = (float*)(smem + 164864);   // final f32 Al tile (32 KB)

    // init Al[b,l] = mps[0,0][l,0]*cos + mps[0,0][l,1]*sin (site 0)
    {
      int m = tid >> 4;
      int lb = (tid & 15) << 3;
      float sn, cs;
      __sincosf(PI_2 * x[(size_t)(rowbase + m) * Nn + 0], &sn, &cs);
      const float* m00 = mps;
#pragma unroll
      for (int j = 0; j < 8; ++j) {
        int l = lb + j;
        float val = m00[2 * l] * cs + m00[2 * l + 1] * sn;
        ALhi[m * Dd + l] = bf_hi(val);
        ALlo[m * Dd + l] = bf_lo(val);
      }
    }
    stage_site(mps + (size_t)1 * SITE_FLOATS, Thi, Tlo, tid);
    if (tid < MAIN_ROWS) {
      float* e1 = eBuf + (1 & 1) * 128;
      __sincosf(PI_2 * x[(size_t)(rowbase + tid) * Nn + 1], &e1[64 + tid], &e1[tid]);
    }
    __syncthreads();

    const int mt = wave >> 3;   // 4 M-tiles (64 rows)
    const int nt = wave & 7;    // 8 N-tiles (128 cols)
    for (int n = 1; n <= Nn - 2; ++n) {
      v8f acc0 = vzero(), acc1 = vzero();
#pragma unroll
      for (int ks = 0; ks < 4; ++ks) {
        const __bf16* pa  = ALhi + (mt * 16 + l15) * Dd + ks * 32 + hlf * 16;
        const __bf16* pal = ALlo + (mt * 16 + l15) * Dd + ks * 32 + hlf * 16;
        v16bf ah = fragA(pa), aL = fragA(pal);  // shared across both channels
        const __bf16* pb0  = Thi + (nt * 16 + l15) * Dd + ks * 32 + hlf * 16;
        const __bf16* pb0l = Tlo + (nt * 16 + l15) * Dd + ks * 32 + hlf * 16;
        acc0 = wmma3(ah, aL, fragB(pb0), fragB(pb0l), acc0);
        acc1 = wmma3(ah, aL, fragB(pb0 + Dd * Dd), fragB(pb0l + Dd * Dd), acc1);
      }
      __syncthreads();

      // fused phase: elementwise combine + AL writeback, then next-site staging
      const float* eC = eBuf + (n & 1) * 128;
      const float* eS = eC + 64;
      int ncol = nt * 16 + l15;
      int mb = mt * 16 + hlf * 8;
      if (n < Nn - 2) {
#pragma unroll
        for (int v2 = 0; v2 < 8; ++v2) {
          int m = mb + v2;
          float val = eC[m] * acc0[v2] + eS[m] * acc1[v2];
          ALhi[m * Dd + ncol] = bf_hi(val);
          ALlo[m * Dd + ncol] = bf_lo(val);
        }
      } else {
#pragma unroll
        for (int v2 = 0; v2 < 8; ++v2) {
          int m = mb + v2;
          Alf[m * Dd + ncol] = eC[m] * acc0[v2] + eS[m] * acc1[v2];
        }
      }
      if (n + 1 <= Nn - 2) {
        stage_site(mps + (size_t)(n + 1) * SITE_FLOATS, Thi, Tlo, tid);
        if (tid < MAIN_ROWS) {
          float* eN = eBuf + ((n + 1) & 1) * 128;
          __sincosf(PI_2 * x[(size_t)(rowbase + tid) * Nn + (n + 1)], &eN[64 + tid], &eN[tid]);
        }
        if (n + 2 <= Nn - 2)
          __builtin_prefetch((const char*)(mps + (size_t)(n + 2) * SITE_FLOATS) + tid * 128, 0, 1);
      }
      __syncthreads();
    }

    // amp[b] = sum_l Al[b,l] * (cos*An[l,0] + sin*An[l,1]),  An = mps[-1,:,0,:]
    if (tid < MAIN_ROWS) {
      int b = rowbase + tid;
      float sn, cs;
      __sincosf(PI_2 * x[(size_t)b * Nn + (Nn - 1)], &sn, &cs);
      const float* last = mps + (size_t)(Nn - 1) * SITE_FLOATS;
      float amp = 0.f;
#pragma unroll 4
      for (int l = 0; l < Dd; ++l) {
        float Ar = cs * last[(size_t)l * Dd * 2 + 0] + sn * last[(size_t)l * Dd * 2 + 1];
        amp += Alf[tid * Dd + l] * Ar;
      }
      ws[b] = amp;
    }
  }
}

__global__ void finalize_kernel(const float* __restrict__ ws, float* __restrict__ out) {
  int b = blockIdx.x * blockDim.x + threadIdx.x;
  if (b < Bb) {
    float a = ws[b];
    out[b] = (a * a) / ws[Bb];
  }
}

extern "C" void kernel_launch(void* const* d_in, const int* in_sizes, int n_in,
                              void* d_out, int out_size, void* d_ws, size_t ws_size,
                              hipStream_t stream) {
  (void)in_sizes; (void)n_in; (void)out_size; (void)ws_size;
  const float* x   = (const float*)d_in[0];   // (B, N) f32
  const float* mps = (const float*)d_in[1];   // (N, D, D, 2) f32
  float* ws  = (float*)d_ws;                  // [0..511]=amp, [512]=norm^2
  float* out = (float*)d_out;                 // (B,) f32

  // block 0: norm chain; blocks 1..8: 64-row batch tiles. 256 KB dynamic LDS.
  mps_kernel<<<dim3(1 + MAIN_BLOCKS), dim3(1024), 262144, stream>>>(x, mps, ws);
  finalize_kernel<<<dim3((Bb + 255) / 256), dim3(256), 0, stream>>>(ws, out);
}
